// SNNBlock_65214783422961
// MI455X (gfx1250) — matrix operand order repaired and checked by reading, not compiled, not run on hardware
//
#include <hip/hip_runtime.h>

// SNN block for MI455X (gfx1250, wave32):
//   Phase 1: split f32 x and W into bf16 (hi, lo) pairs in d_ws   (one pass, ~84 MB ws)
//   Phase 2: fused persistent kernel. Each wave owns a 32x32 (B x H) tile
//            (4 WMMA accumulator tiles), carries the LIF membrane state in
//            VGPRs across T=64 steps, and per step runs the K=2048 GEMM as
//            64 chunks x 12 x v_wmma_f32_16x16x32_bf16 (bf16x3 split
//            precision ~= fp32 accuracy at bf16 matrix throughput).
//   All global accesses are written as uniform-base + 32-bit lane offset so
//   the backend selects SADDR+voffset addressing: no per-chunk VALU address
//   math, K folded into the 24-bit immediate, no WMMA->VALU hazard NOPs.
//   The 4 waves of a block stack along M (batch) at the same H columns, so
//   they read identical W cache lines -> WGP$ absorbs the redundancy.
//   Outputs (134 MB, write-once) use non-temporal stores so the hot
//   W/x split working set (~84 MB) stays resident in the 192 MB L2.

#define T_STEPS 64
#define B_SZ    128
#define I_SZ    2048
#define H_SZ    2048
#define BETA    0.9f
#define THRESH  1.0f

typedef __attribute__((ext_vector_type(16))) __bf16 v16bf;
typedef __attribute__((ext_vector_type(8)))  __bf16 v8bf;
typedef __attribute__((ext_vector_type(8)))  float  v8f;

static __device__ __forceinline__ v16bf pack16(v8bf lo8, v8bf hi8) {
    union { v16bf v; v8bf h[2]; } u;
    u.h[0] = lo8;
    u.h[1] = hi8;
    return u.v;
}

// ---------------- Phase 1: f32 -> (bf16 hi, bf16 lo) split ----------------
__global__ void split_bf16x2_kernel(const float* __restrict__ src,
                                    __bf16* __restrict__ hi,
                                    __bf16* __restrict__ lo,
                                    int n) {
    int i      = blockIdx.x * blockDim.x + threadIdx.x;
    int stride = gridDim.x * blockDim.x;
    for (; i < n; i += stride) {
        float v  = __builtin_nontemporal_load(src + i);   // read-once f32 stream
        __bf16 h = (__bf16)v;
        float  r = v - (float)h;
        hi[i] = h;                                        // hot: keep cacheable
        lo[i] = (__bf16)r;
    }
}

// ---------------- Phase 2: fused WMMA GEMM + LIF scan ----------------
__global__ __launch_bounds__(128) void snn_wmma_scan_kernel(
    const __bf16* __restrict__ xhi, const __bf16* __restrict__ xlo,
    const __bf16* __restrict__ whi, const __bf16* __restrict__ wlo,
    const float*  __restrict__ bias,
    float* __restrict__ spk_out, float* __restrict__ mem_out)
{
    const int lane = threadIdx.x & 31;
    const int wave = threadIdx.x >> 5;
    const int m0   = wave << 5;          // 4 waves stack along M: rows m0..m0+31
    const int h0   = blockIdx.x << 5;    // 64 blocks along H: cols h0..h0+31
    const int half = lane >> 4;          // 0: lanes 0-15, 1: lanes 16-31
    const int ln   = lane & 15;

    // Loop-invariant 32-bit per-lane element offsets (single VGPR each).
    // A fragment f: row m0+16f+ln, K phase half*16 (contiguous, matching B:
    //   lanes 0-15 hold K{0-15}, lanes 16-31 hold K{16-31}).
    // B fragment g: column h0+16g+ln, K phase half*16 (contiguous 32B run).
    uint32_t aoff[2], woff[2];
#pragma unroll
    for (int f = 0; f < 2; ++f)
        aoff[f] = (uint32_t)(m0 + 16 * f + ln) * I_SZ + (uint32_t)half * 16;
#pragma unroll
    for (int g = 0; g < 2; ++g)
        woff[g] = (uint32_t)(h0 + 16 * g + ln) * I_SZ + (uint32_t)half * 16;

    float biasN[2];
#pragma unroll
    for (int g = 0; g < 2; ++g) biasN[g] = bias[h0 + 16 * g + ln];

    v8f mem[2][2];
#pragma unroll
    for (int f = 0; f < 2; ++f)
#pragma unroll
        for (int g = 0; g < 2; ++g)
            mem[f][g] = (v8f){0.f, 0.f, 0.f, 0.f, 0.f, 0.f, 0.f, 0.f};

    for (int t = 0; t < T_STEPS; ++t) {
        // Uniform (SGPR) bases for this timestep.
        const __bf16* axh = xhi + (size_t)t * (B_SZ * I_SZ);
        const __bf16* axl = xlo + (size_t)t * (B_SZ * I_SZ);

        v8f acc[2][2];
#pragma unroll
        for (int f = 0; f < 2; ++f)
#pragma unroll
            for (int g = 0; g < 2; ++g)
                acc[f][g] = (v8f){0.f, 0.f, 0.f, 0.f, 0.f, 0.f, 0.f, 0.f};

        for (int k0 = 0; k0 < I_SZ; k0 += 32) {
            // SADDR form: base(SGPR) + aoff/woff(VGPR) + k0 (imm offset).
            v16bf ah[2], al[2];
#pragma unroll
            for (int f = 0; f < 2; ++f) {
                ah[f] = pack16(*(const v8bf*)(axh + aoff[f] + k0),
                               *(const v8bf*)(axh + aoff[f] + k0 + 8));
                al[f] = pack16(*(const v8bf*)(axl + aoff[f] + k0),
                               *(const v8bf*)(axl + aoff[f] + k0 + 8));
            }
            v16bf bh[2], bl[2];
#pragma unroll
            for (int g = 0; g < 2; ++g) {
                bh[g] = *(const v16bf*)(whi + woff[g] + k0);
                bl[g] = *(const v16bf*)(wlo + woff[g] + k0);
            }
            // bf16x3: hi*hi + hi*lo + lo*hi, f32 accumulate. 12 WMMAs reuse
            // 4 A-fragments and 4 B-fragments (2x fragment reuse each).
#pragma unroll
            for (int f = 0; f < 2; ++f) {
#pragma unroll
                for (int g = 0; g < 2; ++g) {
                    acc[f][g] = __builtin_amdgcn_wmma_f32_16x16x32_bf16(
                        false, ah[f], false, bh[g], (short)0, acc[f][g], false, false);
                    acc[f][g] = __builtin_amdgcn_wmma_f32_16x16x32_bf16(
                        false, ah[f], false, bl[g], (short)0, acc[f][g], false, false);
                    acc[f][g] = __builtin_amdgcn_wmma_f32_16x16x32_bf16(
                        false, al[f], false, bh[g], (short)0, acc[f][g], false, false);
                }
            }
        }

        // C/D layout: VGPR r holds (M = 16f + half*8 + r, N = 16g + ln).
        // LIF update is elementwise per lane; bias depends only on column.
        // Stores: uniform per-t base + 32-bit lane index (SADDR form again).
        float* spk_base = spk_out + (size_t)t * (B_SZ * H_SZ);
        float* mem_base = mem_out + (size_t)t * (B_SZ * H_SZ);
#pragma unroll
        for (int f = 0; f < 2; ++f) {
#pragma unroll
            for (int g = 0; g < 2; ++g) {
                const uint32_t obase =
                    (uint32_t)(m0 + 16 * f + half * 8) * H_SZ +
                    (uint32_t)(h0 + 16 * g + ln);
#pragma unroll
                for (int r = 0; r < 8; ++r) {
                    float cur   = acc[f][g][r] + biasN[g];
                    float mprev = mem[f][g][r];
                    float reset = (mprev > THRESH) ? THRESH : 0.0f;
                    float mnew  = BETA * mprev + cur - reset;
                    float spk   = (mnew > THRESH) ? 1.0f : 0.0f;
                    mem[f][g][r] = mnew;
                    // Write-once streams: NT stores keep W/x resident in L2.
                    __builtin_nontemporal_store(spk, spk_base + obase + r * H_SZ);
                    __builtin_nontemporal_store(mnew, mem_base + obase + r * H_SZ);
                }
            }
        }
    }
}

extern "C" void kernel_launch(void* const* d_in, const int* in_sizes, int n_in,
                              void* d_out, int out_size, void* d_ws, size_t ws_size,
                              hipStream_t stream) {
    const float* x = (const float*)d_in[0];   // [T, B, I]
    const float* W = (const float*)d_in[1];   // [H, I]
    const float* b = (const float*)d_in[2];   // [H]

    const size_t XN = (size_t)T_STEPS * B_SZ * I_SZ;  // 16,777,216
    const size_t WN = (size_t)H_SZ * I_SZ;            //  4,194,304
    // Workspace layout (bf16): x_hi | x_lo | w_hi | w_lo  => ~84 MB total.
    __bf16* xhi = (__bf16*)d_ws;
    __bf16* xlo = xhi + XN;
    __bf16* whi = xlo + XN;
    __bf16* wlo = whi + WN;

    split_bf16x2_kernel<<<2048, 256, 0, stream>>>(x, xhi, xlo, (int)XN);
    split_bf16x2_kernel<<<512, 256, 0, stream>>>(W, whi, wlo, (int)WN);

    float* spk  = (float*)d_out;
    float* memo = spk + (size_t)T_STEPS * B_SZ * H_SZ;

    // 256 wave-tiles of 32x32: 64 blocks x 4 waves; waves stack along M so a
    // block's waves share W cache lines (WGP$), blocks sweep H.
    snn_wmma_scan_kernel<<<64, 128, 0, stream>>>(xhi, xlo, whi, wlo, b, spk, memo);
}
